// SwinBlock_4337916970078
// MI455X (gfx1250) — compile-verified, matrix operand order checked by reading, not run on hardware
//
#include <hip/hip_runtime.h>
#include <cstdint>
#include <cstddef>

typedef __attribute__((ext_vector_type(16))) _Float16 v16h;
typedef __attribute__((ext_vector_type(8)))  float    v8f;
typedef unsigned int v4u __attribute__((ext_vector_type(4)));
typedef int          v8i __attribute__((ext_vector_type(8)));
typedef int          v4i __attribute__((ext_vector_type(4)));

#if defined(__has_builtin)
#if __has_builtin(__builtin_amdgcn_tensor_load_to_lds)
#define HAVE_TDM 1
#endif
#endif
#ifndef HAVE_TDM
#define HAVE_TDM 0
#endif

constexpr int kDim   = 192;
constexpr int kHeads = 6;
constexpr int kHd    = 32;
constexpr int kWin   = 8;     // window size
constexpr int kShift = 4;
constexpr int kHid   = 768;
constexpr int kB     = 8;
constexpr int kH     = 128;
constexpr int kW     = 128;
constexpr int kL     = kH * kW;      // 16384
constexpr int kTok   = kB * kL;      // 131072
constexpr int kNWin  = (kH / kWin) * (kW / kWin);  // 256 windows per image
constexpr int kN     = kWin * kWin;  // 64 tokens per window

// ---------------------------------------------------------------------------
// helpers
// ---------------------------------------------------------------------------
__device__ __forceinline__ float gelu_f(float x) {
  return 0.5f * x * (1.0f + erff(x * 0.70710678118654752f));
}

// A/B fragment gather from row-major [rows x K] f16 (element (row0+m, k0+K)).
// Per ISA: lanes 0-15 hold rows with K in {0..7,16..23}, lanes 16-31 same rows
// with K in {8..15,24..31}.
__device__ __forceinline__ v16h frag_rm(const _Float16* p, int ld, int lane) {
  const int m = lane & 15, half = lane >> 4;
  const _Float16* q = p + (size_t)m * ld + (half << 3);
  v16h f;
#pragma unroll
  for (int i = 0; i < 16; ++i) {
    const int K = (i & 7) + ((i >> 3) << 4);
    f[i] = q[K];
  }
  return f;
}

// B fragment gather from K-major memory [K x N] (element (k0+K, n0+n)).
__device__ __forceinline__ v16h frag_cm(const _Float16* p, int ld, int lane) {
  const int n = lane & 15, half = lane >> 4;
  const _Float16* q = p + n + (size_t)(half << 3) * ld;
  v16h f;
#pragma unroll
  for (int i = 0; i < 16; ++i) {
    const int K = (i & 7) + ((i >> 3) << 4);
    f[i] = q[(size_t)K * ld];
  }
  return f;
}

// Issue a TDM DMA of a [rows x 32] f16 tile (row stride = ldK elements) from
// global memory into LDS. Descriptor layout per CDNA5 ISA ch.8 (D# groups).
__device__ __forceinline__ void tdm_load_tile_f16(const _Float16* gsrc,
                                                  void* lds_dst, int ldK,
                                                  int rows) {
#if HAVE_TDM
  const uint64_t ga = (uint64_t)(uintptr_t)gsrc;
  v4u g0;
  g0[0] = 1u;                                   // count=1, user mode, no gather
  g0[1] = (unsigned)(uintptr_t)lds_dst;         // lds_addr
  g0[2] = (unsigned)ga;                         // global_addr[31:0]
  g0[3] = (unsigned)((ga >> 32) & 0x1FFFFFFu) | (2u << 30);  // addr[56:32]|type=2
  v8i g1;
  g1[0] = (int)(1u << 16);                      // wg_mask=0, data_size=2B
  g1[1] = (int)(((unsigned)ldK & 0xFFFFu) << 16);            // tensor_dim0 lo16
  g1[2] = (int)((((unsigned)ldK >> 16) & 0xFFFFu) |
                (((unsigned)rows & 0xFFFFu) << 16));         // dim0 hi | dim1 lo
  g1[3] = (int)(32u << 16);                     // dim1 hi=0 | tile_dim0=32
  g1[4] = rows;                                 // tile_dim1=rows, tile_dim2=0
  g1[5] = ldK;                                  // tensor_dim0_stride lo32
  g1[6] = 0;                                    // stride hi16 | dim1_stride lo16
  g1[7] = 0;                                    // dim1_stride hi32
  v4i z4 = {};
#if defined(__clang_major__) && (__clang_major__ >= 23)
  v8i z8 = {};
  __builtin_amdgcn_tensor_load_to_lds(g0, g1, z4, z4, z8, 0);
#else
  __builtin_amdgcn_tensor_load_to_lds(g0, g1, z4, z4, 0);
#endif
  __builtin_amdgcn_s_wait_tensorcnt(0);
#else
  (void)gsrc; (void)lds_dst; (void)ldK; (void)rows;
#endif
}

// window-ordered row index -> original token index (window_reverse + roll(+4))
__device__ __forceinline__ int winrow_to_token(int r) {
  const int wg = r >> 6, t = r & 63;
  const int b = wg >> 8, wr = wg & 255;
  const int wh = wr >> 4, ww = wr & 15;
  const int a = wh * kWin + (t >> 3);
  const int c = ww * kWin + (t & 7);
  const int h = (a + kShift) & (kH - 1);
  const int w = (c + kShift) & (kW - 1);
  return (b << 14) + (h << 7) + w;
}

// original token index -> window-ordered row (roll(-4) + window_partition)
__device__ __forceinline__ int token_to_winrow(int t) {
  const int b = t >> 14, hw = t & (kL - 1);
  const int h = hw >> 7, w = hw & (kW - 1);
  const int a = (h - kShift) & (kH - 1);
  const int c = (w - kShift) & (kW - 1);
  const int wg = (b << 8) + (a >> 3) * 16 + (c >> 3);
  return (wg << 6) + (a & 7) * kWin + (c & 7);
}

// region id for the shift-window attention mask
__device__ __forceinline__ int region_id(int h, int w) {
  const int rh = (h < kH - kWin) ? 0 : (h < kH - kShift ? 1 : 2);
  const int rw = (w < kW - kWin) ? 0 : (w < kW - kShift ? 1 : 2);
  return rh * 3 + rw;
}

// ---------------------------------------------------------------------------
// small prep kernels
// ---------------------------------------------------------------------------
__global__ void f32_to_f16_kernel(const float* __restrict__ in,
                                  _Float16* __restrict__ out, int n) {
  const int i = blockIdx.x * 256 + threadIdx.x;
  if (i < n) out[i] = (_Float16)in[i];
}

// OIHW [CO][CI][3][3] f32 -> tap-major [9][CO][CI] f16
__global__ void conv_w_relayout_kernel(const float* __restrict__ in,
                                       _Float16* __restrict__ out, int CO, int CI) {
  const int i = blockIdx.x * 256 + threadIdx.x;
  const int n = CO * CI * 9;
  if (i >= n) return;
  const int tap = i / (CO * CI);
  const int rem = i % (CO * CI);
  const int o = rem / CI, ci = rem % CI;
  out[i] = (_Float16)in[(size_t)(o * CI + ci) * 9 + tap];
}

__global__ void qkvbias_kernel(const float* __restrict__ qb,
                               const float* __restrict__ vb,
                               float* __restrict__ out) {
  const int i = blockIdx.x * 256 + threadIdx.x;
  if (i < kDim) out[i] = qb[i];
  else if (i < 2 * kDim) out[i] = 0.0f;
  else if (i < 3 * kDim) out[i] = vb[i - 2 * kDim];
}

// CPB MLP -> rpb16[h][query][key] = 16*sigmoid(bias), single block
__global__ void rpb_kernel(const float* __restrict__ cw1, const float* __restrict__ cb1,
                           const float* __restrict__ cw2,
                           float* __restrict__ rpb16, float* __restrict__ hbias) {
  const int T = (2 * kWin - 1) * (2 * kWin - 1);  // 225
  for (int idx = threadIdx.x; idx < T; idx += 256) {
    const int iy = idx / 15, ix = idx % 15;
    float ty = (float)(iy - 7) / 7.0f * 8.0f;
    float tx = (float)(ix - 7) / 7.0f * 8.0f;
    ty = copysignf(log2f(fabsf(ty) + 1.0f) * (1.0f / 3.0f), ty);
    tx = copysignf(log2f(fabsf(tx) + 1.0f) * (1.0f / 3.0f), tx);
    float acc[kHeads] = {};
    for (int k = 0; k < 512; ++k) {
      float hv = cw1[k * 2 + 0] * ty + cw1[k * 2 + 1] * tx + cb1[k];
      hv = fmaxf(hv, 0.0f);
#pragma unroll
      for (int h = 0; h < kHeads; ++h) acc[h] += cw2[h * 512 + k] * hv;
    }
#pragma unroll
    for (int h = 0; h < kHeads; ++h) hbias[idx * kHeads + h] = acc[h];
  }
  __syncthreads();
  for (int e = threadIdx.x; e < kHeads * kN * kN; e += 256) {
    const int h = e / (kN * kN), nm = e % (kN * kN);
    const int q = nm >> 6, m = nm & 63;
    const int dy = (q >> 3) - (m >> 3) + 7;
    const int dx = (q & 7) - (m & 7) + 7;
    const float v = hbias[(dy * 15 + dx) * kHeads + h];
    rpb16[e] = 16.0f / (1.0f + expf(-v));
  }
}

// ---------------------------------------------------------------------------
// LayerNorm: one wave per token (wave32), writes f16 token-order and
// (optionally) f16 window-gathered order.
// ---------------------------------------------------------------------------
__global__ __launch_bounds__(256) void ln_kernel(
    const float* __restrict__ x, const float* __restrict__ w,
    const float* __restrict__ b, _Float16* __restrict__ out_plain,
    _Float16* __restrict__ out_win) {
  const int tok = blockIdx.x * 8 + (threadIdx.x >> 5);
  const int lane = threadIdx.x & 31;
  const float* xp = x + (size_t)tok * kDim;
  float v[6];
  float s = 0.0f;
#pragma unroll
  for (int i = 0; i < 6; ++i) { v[i] = xp[lane + 32 * i]; s += v[i]; }
#pragma unroll
  for (int o = 16; o > 0; o >>= 1) s += __shfl_xor(s, o, 32);
  const float mu = s * (1.0f / (float)kDim);
  float q = 0.0f;
#pragma unroll
  for (int i = 0; i < 6; ++i) { const float d = v[i] - mu; q += d * d; }
#pragma unroll
  for (int o = 16; o > 0; o >>= 1) q += __shfl_xor(q, o, 32);
  const float rstd = rsqrtf(q * (1.0f / (float)kDim) + 1e-5f);
  const int wr = token_to_winrow(tok);
#pragma unroll
  for (int i = 0; i < 6; ++i) {
    const int c = lane + 32 * i;
    const float y = (v[i] - mu) * rstd * w[c] + b[c];
    const _Float16 hy = (_Float16)y;
    if (out_plain) out_plain[(size_t)tok * kDim + c] = hy;
    if (out_win) out_win[(size_t)wr * kDim + c] = hy;
  }
}

// ---------------------------------------------------------------------------
// Generic WMMA GEMM: Y[M,N] = A[M,K] * W[N,K]^T + bias, 128x64 block tile,
// 8 waves. Weight tile staged in LDS via the Tensor Data Mover (wave 0 issues
// the DMA, waits TENSORcnt, barrier releases the block); cooperative-load
// fallback if the TDM builtin is unavailable.
// ---------------------------------------------------------------------------
enum GemmEpi { EPI_F16 = 0, EPI_F16_GELU = 1, EPI_PROJ = 2, EPI_RESID = 3 };

template <int EPI>
__global__ __launch_bounds__(256) void gemm_wmma_kernel(
    const _Float16* __restrict__ A, const _Float16* __restrict__ W,
    const float* __restrict__ bias, int K, int N, _Float16* __restrict__ outh,
    float* __restrict__ outf, const float* __restrict__ resid) {
  __shared__ _Float16 sW[64][32];
  const int lane = threadIdx.x & 31;
  const int wave = threadIdx.x >> 5;
  const int m0 = blockIdx.x * 128 + wave * 16;
  const int n0 = blockIdx.y * 64;
  v8f acc[4] = {};
  for (int k0 = 0; k0 < K; k0 += 32) {
#if HAVE_TDM
    if (threadIdx.x < 32) {
      tdm_load_tile_f16(W + (size_t)n0 * K + k0, &sW[0][0], K, 64);
    }
#else
    for (int e = threadIdx.x; e < 64 * 32; e += 256) {
      const int n = e >> 5, kk = e & 31;
      sW[n][kk] = W[(size_t)(n0 + n) * K + (k0 + kk)];
    }
#endif
    __syncthreads();
    if (k0 + 32 < K) __builtin_prefetch(A + (size_t)m0 * K + k0 + 32, 0, 1);
    const v16h a = frag_rm(A + (size_t)m0 * K + k0, K, lane);
#pragma unroll
    for (int nt = 0; nt < 4; ++nt) {
      const v16h b = frag_rm(&sW[nt * 16][0], 32, lane);
      acc[nt] = __builtin_amdgcn_wmma_f32_16x16x32_f16(false, a, false, b,
                                                       (short)0, acc[nt],
                                                       false, false);
    }
    __syncthreads();
  }
  const int nl = lane & 15;
  const int mh = (lane >> 4) * 8;
#pragma unroll
  for (int nt = 0; nt < 4; ++nt) {
    const int n = n0 + nt * 16 + nl;
    const float bv = bias[n];
#pragma unroll
    for (int r = 0; r < 8; ++r) {
      const int m = m0 + mh + r;
      const float v = acc[nt][r] + bv;
      if (EPI == EPI_F16) {
        outh[(size_t)m * N + n] = (_Float16)v;
      } else if (EPI == EPI_F16_GELU) {
        outh[(size_t)m * N + n] = (_Float16)gelu_f(v);
      } else if (EPI == EPI_PROJ) {
        outf[(size_t)winrow_to_token(m) * N + n] = v;  // window_reverse+roll
      } else {
        const size_t idx = (size_t)m * N + n;
        outf[idx] = v + resid[idx];
      }
    }
  }
}

// ---------------------------------------------------------------------------
// Shifted-window cosine attention: one block per (window, head).
// ---------------------------------------------------------------------------
__global__ __launch_bounds__(256) void attn_kernel(
    const _Float16* __restrict__ qkv, const float* __restrict__ logit_scale,
    const float* __restrict__ rpb16, _Float16* __restrict__ outh) {
  __shared__ _Float16 sq[kN][kHd], sk[kN][kHd], sv[kN][kHd];
  __shared__ float sS[kN][kN];
  __shared__ _Float16 sP[kN][kN];
  __shared__ float rq[kN], rk[kN];

  const int blk = blockIdx.x;
  const int win = blk / kHeads, hd = blk % kHeads;
  const int lane = threadIdx.x & 31, wave = threadIdx.x >> 5;
  const _Float16* base = qkv + (size_t)win * kN * (3 * kDim) + hd * kHd;

  for (int e = threadIdx.x; e < kN * kHd; e += 256) {
    const int t = e >> 5, d = e & 31;
    sq[t][d] = base[(size_t)t * (3 * kDim) + d];
    sk[t][d] = base[(size_t)t * (3 * kDim) + kDim + d];
    sv[t][d] = base[(size_t)t * (3 * kDim) + 2 * kDim + d];
  }
  __syncthreads();
  if (threadIdx.x < kN) {
    const int t = threadIdx.x;
    float a2 = 0.0f, b2 = 0.0f;
    for (int d = 0; d < kHd; ++d) {
      const float a = (float)sq[t][d]; a2 += a * a;
      const float b = (float)sk[t][d]; b2 += b * b;
    }
    rq[t] = 1.0f / fmaxf(sqrtf(a2), 1e-12f);
    rk[t] = 1.0f / fmaxf(sqrtf(b2), 1e-12f);
  }
  __syncthreads();
  for (int e = threadIdx.x; e < kN * kHd; e += 256) {
    const int t = e >> 5, d = e & 31;
    sq[t][d] = (_Float16)((float)sq[t][d] * rq[t]);
    sk[t][d] = (_Float16)((float)sk[t][d] * rk[t]);
  }
  __syncthreads();

  const float scale = expf(fminf(logit_scale[hd], 4.605170186f));  // ln(100)
  const int wr = win & 255, wh = wr >> 4, ww = wr & 15;

  // S = qn * kn^T  (4x4 tiles of 16x16, K=32: one WMMA each; 2 tiles/wave)
  for (int tile = wave; tile < 16; tile += 8) {
    const int mi = tile >> 2, ni = tile & 3;
    const v16h a = frag_rm(&sq[mi * 16][0], kHd, lane);
    const v16h b = frag_rm(&sk[ni * 16][0], kHd, lane);
    v8f c = {};
    c = __builtin_amdgcn_wmma_f32_16x16x32_f16(false, a, false, b, (short)0, c,
                                               false, false);
    const int n = ni * 16 + (lane & 15);   // key index
    const int mh = (lane >> 4) * 8;
    const int regk = region_id(wh * kWin + (n >> 3), ww * kWin + (n & 7));
#pragma unroll
    for (int r = 0; r < 8; ++r) {
      const int m = mi * 16 + mh + r;      // query index
      const int regq = region_id(wh * kWin + (m >> 3), ww * kWin + (m & 7));
      float v = c[r] * scale + rpb16[(hd << 12) + (m << 6) + n];
      if (regq != regk) v -= 100.0f;
      sS[m][n] = v;
    }
  }
  __syncthreads();

  // softmax over keys
  if (threadIdx.x < kN) {
    const int m = threadIdx.x;
    float mx = -1e30f;
    for (int n = 0; n < kN; ++n) mx = fmaxf(mx, sS[m][n]);
    float sum = 0.0f;
    for (int n = 0; n < kN; ++n) {
      const float e = expf(sS[m][n] - mx);
      sS[m][n] = e;
      sum += e;
    }
    const float inv = 1.0f / sum;
    for (int n = 0; n < kN; ++n) sP[m][n] = (_Float16)(sS[m][n] * inv);
  }
  __syncthreads();

  // O = P * V  (4x2 tiles, K=64: two WMMA each; 1 tile/wave)
  {
    const int mi = wave >> 1, ni = wave & 1;
    v8f c = {};
#pragma unroll
    for (int ks = 0; ks < 2; ++ks) {
      const v16h a = frag_rm(&sP[mi * 16][ks * 32], kN, lane);
      const v16h b = frag_cm(&sv[ks * 32][ni * 16], kHd, lane);
      c = __builtin_amdgcn_wmma_f32_16x16x32_f16(false, a, false, b, (short)0,
                                                 c, false, false);
    }
    const int n = ni * 16 + (lane & 15);
    const int mh = (lane >> 4) * 8;
#pragma unroll
    for (int r = 0; r < 8; ++r) {
      const int m = mi * 16 + mh + r;
      outh[((size_t)win * kN + m) * kDim + hd * kHd + n] = (_Float16)c[r];
    }
  }
}

// ---------------------------------------------------------------------------
// 3x3 conv as implicit WMMA GEMM over NHWC f16. One block per image row,
// 8 waves x 16 pixels, tap-major weights staged in LDS (cooperative path).
// ---------------------------------------------------------------------------
template <int CIN>
__device__ __forceinline__ v16h frag_conv_a(const _Float16* base, int w0,
                                            int dw, int k0, int lane) {
  const int m = lane & 15, half = lane >> 4;
  const int col = w0 + m + dw;
  v16h f;
  if (col < 0 || col >= kW) {
#pragma unroll
    for (int i = 0; i < 16; ++i) f[i] = (_Float16)0.0f;
    return f;
  }
  const _Float16* q = base + (size_t)col * CIN + k0 + (half << 3);
#pragma unroll
  for (int i = 0; i < 16; ++i) f[i] = q[(i & 7) + ((i >> 3) << 4)];
  return f;
}

template <int CIN, int COUT, bool GACT, bool F32OUT>
__global__ __launch_bounds__(256) void conv3x3_wmma_kernel(
    const _Float16* __restrict__ X, const _Float16* __restrict__ Wt,  // [9][COUT][CIN]
    const float* __restrict__ bias, _Float16* __restrict__ outh,
    float* __restrict__ outf) {
  __shared__ _Float16 sW[64][32];
  const int lane = threadIdx.x & 31, wave = threadIdx.x >> 5;
  const int bh = blockIdx.x;
  const int b = bh >> 7, h = bh & (kH - 1);
  const int w0 = wave * 16;
  const int n0 = blockIdx.y * 64;
  v8f acc[4] = {};
  for (int dh = -1; dh <= 1; ++dh) {
    const int hh = h + dh;
    if (hh < 0 || hh >= kH) continue;  // uniform across block
    const _Float16* base = X + (size_t)(b * kH + hh) * kW * CIN;
    for (int dw = -1; dw <= 1; ++dw) {
      const int tap = (dh + 1) * 3 + (dw + 1);
      const _Float16* wt = Wt + (size_t)tap * COUT * CIN;
      for (int k0 = 0; k0 < CIN; k0 += 32) {
        for (int e = threadIdx.x; e < 64 * 32; e += 256) {
          const int n = e >> 5, kk = e & 31;
          sW[n][kk] = wt[(size_t)(n0 + n) * CIN + (k0 + kk)];
        }
        __syncthreads();
        const v16h a = frag_conv_a<CIN>(base, w0, dw, k0, lane);
#pragma unroll
        for (int nt = 0; nt < 4; ++nt) {
          const v16h bb = frag_rm(&sW[nt * 16][0], 32, lane);
          acc[nt] = __builtin_amdgcn_wmma_f32_16x16x32_f16(
              false, a, false, bb, (short)0, acc[nt], false, false);
        }
        __syncthreads();
      }
    }
  }
  const int nl = lane & 15;
  const int mh = (lane >> 4) * 8;
  const size_t pixbase = (size_t)(b * kH + h) * kW;
#pragma unroll
  for (int nt = 0; nt < 4; ++nt) {
    const int n = n0 + nt * 16 + nl;
    const float bv = bias[n];
#pragma unroll
    for (int r = 0; r < 8; ++r) {
      const int px = w0 + mh + r;
      float v = acc[nt][r] + bv;
      if (GACT) v = gelu_f(v);
      if (F32OUT) outf[(pixbase + px) * COUT + n] = v;
      else outh[(pixbase + px) * COUT + n] = (_Float16)v;
    }
  }
}

// channel means of conv2 output: one block per (b, c)
__global__ __launch_bounds__(256) void chansum_kernel(
    const float* __restrict__ c2, float* __restrict__ mean) {
  __shared__ float red[256];
  const int b = blockIdx.x / kDim, c = blockIdx.x % kDim;
  float s = 0.0f;
  for (int p = threadIdx.x; p < kL; p += 256)
    s += c2[((size_t)b * kL + p) * kDim + c];
  red[threadIdx.x] = s;
  __syncthreads();
  for (int o = 128; o > 0; o >>= 1) {
    if (threadIdx.x < o) red[threadIdx.x] += red[threadIdx.x + o];
    __syncthreads();
  }
  if (threadIdx.x == 0) mean[blockIdx.x] = red[0] * (1.0f / (float)kL);
}

// channel attention: sigmoid(relu(mean@ca1^T+b1)@ca2^T+b2), one block per image
__global__ void ca_kernel(const float* __restrict__ mean,
                          const float* __restrict__ ca1w, const float* __restrict__ ca1b,
                          const float* __restrict__ ca2w, const float* __restrict__ ca2b,
                          float* __restrict__ sfac) {
  __shared__ float s1[8];
  const int b = blockIdx.x;
  if (threadIdx.x < 6) {  // squeeze channels
    const int j = threadIdx.x;
    float s = ca1b[j];
    for (int c = 0; c < kDim; ++c) s += mean[b * kDim + c] * ca1w[j * kDim + c];
    s1[j] = fmaxf(s, 0.0f);
  }
  __syncthreads();
  const int c = threadIdx.x;
  float s = ca2b[c];
#pragma unroll
  for (int j = 0; j < 6; ++j) s += s1[j] * ca2w[c * 6 + j];
  sfac[b * kDim + c] = 1.0f / (1.0f + expf(-s));
}

// x1 = x + attn_x + 0.01 * conv2 * s
__global__ __launch_bounds__(256) void combine_kernel(
    const float* __restrict__ x, const float* __restrict__ attnx,
    const float* __restrict__ c2, const float* __restrict__ sfac,
    float* __restrict__ x1) {
  const size_t i = (size_t)blockIdx.x * 256 + threadIdx.x;
  const int c = (int)(i % kDim);
  const int b = (int)(i / ((size_t)kL * kDim));
  x1[i] = x[i] + attnx[i] + 0.01f * c2[i] * sfac[b * kDim + c];
}

// ---------------------------------------------------------------------------
// launch
// ---------------------------------------------------------------------------
extern "C" void kernel_launch(void* const* d_in, const int* in_sizes, int n_in,
                              void* d_out, int out_size, void* d_ws,
                              size_t ws_size, hipStream_t stream) {
  (void)in_sizes; (void)n_in; (void)out_size; (void)ws_size;
  const float* x      = (const float*)d_in[0];
  const float* n1w    = (const float*)d_in[3];
  const float* n1b    = (const float*)d_in[4];
  const float* qkvw   = (const float*)d_in[5];
  const float* qb     = (const float*)d_in[6];
  const float* vb     = (const float*)d_in[7];
  const float* lscale = (const float*)d_in[8];
  const float* cpbw1  = (const float*)d_in[9];
  const float* cpbb1  = (const float*)d_in[10];
  const float* cpbw2  = (const float*)d_in[11];
  const float* projw  = (const float*)d_in[12];
  const float* projb  = (const float*)d_in[13];
  const float* cab1w  = (const float*)d_in[14];
  const float* cab1b  = (const float*)d_in[15];
  const float* cab2w  = (const float*)d_in[16];
  const float* cab2b  = (const float*)d_in[17];
  const float* ca1w   = (const float*)d_in[18];
  const float* ca1b   = (const float*)d_in[19];
  const float* ca2w   = (const float*)d_in[20];
  const float* ca2b   = (const float*)d_in[21];
  const float* n2w    = (const float*)d_in[22];
  const float* n2b    = (const float*)d_in[23];
  const float* fc1w   = (const float*)d_in[24];
  const float* fc1b   = (const float*)d_in[25];
  const float* fc2w   = (const float*)d_in[26];
  const float* fc2b   = (const float*)d_in[27];
  float* out = (float*)d_out;

  char* p = (char*)d_ws;
  auto alloc = [&](size_t bytes) -> char* {
    char* r = p;
    p += (bytes + 255) & ~(size_t)255;
    return r;
  };
  _Float16* xh    = (_Float16*)alloc((size_t)kTok * kDim * 2);   // LN1, token order (conv in)
  _Float16* xnw   = (_Float16*)alloc((size_t)kTok * kDim * 2);   // LN1, window order (qkv A)
  _Float16* qkv   = (_Float16*)alloc((size_t)kTok * 3 * kDim * 2);
  _Float16* attno = (_Float16*)alloc((size_t)kTok * kDim * 2);   // attn out, window order
  float*    attnx = (float*)alloc((size_t)kTok * kDim * 4);      // proj out, token order
  _Float16* c1    = (_Float16*)alloc((size_t)kTok * 64 * 2);
  float*    c2    = (float*)alloc((size_t)kTok * kDim * 4);
  float*    x1    = (float*)alloc((size_t)kTok * kDim * 4);
  _Float16* xn2   = (_Float16*)alloc((size_t)kTok * kDim * 2);
  _Float16* hbuf  = (_Float16*)alloc((size_t)kTok * kHid * 2);
  _Float16* wqh   = (_Float16*)alloc((size_t)3 * kDim * kDim * 2);
  _Float16* wph   = (_Float16*)alloc((size_t)kDim * kDim * 2);
  _Float16* wf1h  = (_Float16*)alloc((size_t)kHid * kDim * 2);
  _Float16* wf2h  = (_Float16*)alloc((size_t)kDim * kHid * 2);
  _Float16* wc1h  = (_Float16*)alloc((size_t)9 * 64 * kDim * 2);
  _Float16* wc2h  = (_Float16*)alloc((size_t)9 * kDim * 64 * 2);
  float* qkvbias  = (float*)alloc(3 * kDim * 4);
  float* rpb16    = (float*)alloc((size_t)kHeads * kN * kN * 4);
  float* hbias    = (float*)alloc(225 * kHeads * 4);
  float* cmean    = (float*)alloc((size_t)kB * kDim * 4);
  float* sfac     = (float*)alloc((size_t)kB * kDim * 4);

  const int nqkvw = 3 * kDim * kDim, nprj = kDim * kDim, nfc = kHid * kDim;
  f32_to_f16_kernel<<<(nqkvw + 255) / 256, 256, 0, stream>>>(qkvw, wqh, nqkvw);
  f32_to_f16_kernel<<<(nprj + 255) / 256, 256, 0, stream>>>(projw, wph, nprj);
  f32_to_f16_kernel<<<(nfc + 255) / 256, 256, 0, stream>>>(fc1w, wf1h, nfc);
  f32_to_f16_kernel<<<(nfc + 255) / 256, 256, 0, stream>>>(fc2w, wf2h, nfc);
  conv_w_relayout_kernel<<<(9 * 64 * kDim + 255) / 256, 256, 0, stream>>>(cab1w, wc1h, 64, kDim);
  conv_w_relayout_kernel<<<(9 * kDim * 64 + 255) / 256, 256, 0, stream>>>(cab2w, wc2h, kDim, 64);
  qkvbias_kernel<<<3, 256, 0, stream>>>(qb, vb, qkvbias);
  rpb_kernel<<<1, 256, 0, stream>>>(cpbw1, cpbb1, cpbw2, rpb16, hbias);

  // LN1 (fused shift+window gather)
  ln_kernel<<<kTok / 8, 256, 0, stream>>>(x, n1w, n1b, xh, xnw);
  // QKV GEMM: [131072,192] x [576,192]^T
  gemm_wmma_kernel<EPI_F16><<<dim3(kTok / 128, (3 * kDim) / 64), 256, 0, stream>>>(
      xnw, wqh, qkvbias, kDim, 3 * kDim, qkv, nullptr, nullptr);
  // attention: one block per (window, head)
  attn_kernel<<<kB * kNWin * kHeads, 256, 0, stream>>>(qkv, lscale, rpb16, attno);
  // proj GEMM with fused window_reverse + roll scatter
  gemm_wmma_kernel<EPI_PROJ><<<dim3(kTok / 128, kDim / 64), 256, 0, stream>>>(
      attno, wph, projb, kDim, kDim, nullptr, attnx, nullptr);
  // CAB branch
  conv3x3_wmma_kernel<kDim, 64, true, false><<<dim3(kB * kH, 1), 256, 0, stream>>>(
      xh, wc1h, cab1b, c1, nullptr);
  conv3x3_wmma_kernel<64, kDim, false, true><<<dim3(kB * kH, kDim / 64), 256, 0, stream>>>(
      c1, wc2h, cab2b, nullptr, c2);
  chansum_kernel<<<kB * kDim, 256, 0, stream>>>(c2, cmean);
  ca_kernel<<<kB, kDim, 0, stream>>>(cmean, ca1w, ca1b, ca2w, ca2b, sfac);
  // x1 = x + attn_x + 0.01 * conv_x
  combine_kernel<<<((size_t)kTok * kDim) / 256, 256, 0, stream>>>(x, attnx, c2, sfac, x1);
  // MLP
  ln_kernel<<<kTok / 8, 256, 0, stream>>>(x1, n2w, n2b, xn2, nullptr);
  gemm_wmma_kernel<EPI_F16_GELU><<<dim3(kTok / 128, kHid / 64), 256, 0, stream>>>(
      xn2, wf1h, fc1b, kDim, kHid, hbuf, nullptr, nullptr);
  gemm_wmma_kernel<EPI_RESID><<<dim3(kTok / 128, kDim / 64), 256, 0, stream>>>(
      hbuf, wf2h, fc2b, kHid, kDim, nullptr, out, x1);
}
